// SelfAttention_39324720562709
// MI455X (gfx1250) — compile-verified, matrix operand order checked by reading, not run on hardware
//
#include <hip/hip_runtime.h>

// ---------------------------------------------------------------------------
// SAGAN self-attention for MI455X (gfx1250, wave32, WMMA + TDM).
//   B=4, C=256, N=4096, d=32. Compute-bound (~41 GFLOP vs ~34MB traffic)
//   -> bf16 WMMA everywhere, flash-attention, transposed-S formulation
//   (S^T C/D register layout == bf16 B-matrix layout, so exp(S)->P needs
//   zero cross-lane movement). K tiles are block-shared -> double-buffered
//   Tensor Data Mover loads into LDS with s_wait_tensorcnt.
// ---------------------------------------------------------------------------

typedef __attribute__((ext_vector_type(16))) __bf16 v16bf;
typedef __attribute__((ext_vector_type(8)))  __bf16 v8bf;
typedef __attribute__((ext_vector_type(8)))  float  v8f;
typedef __attribute__((ext_vector_type(4)))  unsigned int v4u;
typedef __attribute__((ext_vector_type(8)))  int v8i;
typedef __attribute__((ext_vector_type(4)))  int v4i;

#define BB 4
#define CC 256
#define DD 32      // C/8
#define NN 4096    // H*W
#define JT 64      // j-tile (keys per iteration)

// ---------------------------------------------------------------------------
// Kernel 1: 1x1-conv projections q,k,v as WMMA GEMMs.
//   grid = (N/16, 20, B), 1 wave/block. g: 0-1 -> q, 2-3 -> k, 4-19 -> v.
//   q,k stored [B,N,32] bf16 (feeds Q B-frags / K tiles),
//   v  stored [B,C,N] bf16 (feeds V A-frags contiguously in j).
// ---------------------------------------------------------------------------
__global__ __launch_bounds__(32) void proj_kernel(
    const float* __restrict__ x,
    const float* __restrict__ Wq, const float* __restrict__ bq,
    const float* __restrict__ Wk, const float* __restrict__ bk,
    const float* __restrict__ Wv, const float* __restrict__ bv,
    unsigned short* __restrict__ qbuf,
    unsigned short* __restrict__ kbuf,
    unsigned short* __restrict__ vbuf)
{
  const int lane = threadIdx.x;
  const int ln = lane & 15;
  const int hi = lane >> 4;
  const int n0 = blockIdx.x * 16;
  const int g  = blockIdx.y;
  const int b  = blockIdx.z;

  const float* W; const float* bias; int o0;
  if (g < 2)      { W = Wq; bias = bq; o0 = g * 16; }
  else if (g < 4) { W = Wk; bias = bk; o0 = (g - 2) * 16; }
  else            { W = Wv; bias = bv; o0 = (g - 4) * 16; }

  const float* xb = x + (size_t)b * CC * NN;
  v8f acc = {};

  for (int c0 = 0; c0 < CC; c0 += 32) {
    // A-frag: rows o0+ln of W; K(c) halves hi*8.. and 16+hi*8..
    v16bf a;
    const float* wr = W + (size_t)(o0 + ln) * CC + c0 + hi * 8;
    #pragma unroll
    for (int e = 0; e < 8; ++e) a[e] = (__bf16)wr[e];
    #pragma unroll
    for (int e = 0; e < 8; ++e) a[8 + e] = (__bf16)wr[16 + e];
    // B-frag: column n0+ln of x (half-wave-coalesced gather)
    v16bf bf;
    const float* xc = xb + (size_t)(c0 + hi * 8) * NN + n0 + ln;
    #pragma unroll
    for (int e = 0; e < 8; ++e) bf[e] = (__bf16)xc[(size_t)e * NN];
    const float* xc2 = xc + (size_t)16 * NN;
    #pragma unroll
    for (int e = 0; e < 8; ++e) bf[8 + e] = (__bf16)xc2[(size_t)e * NN];

    acc = __builtin_amdgcn_wmma_f32_16x16x32_bf16(false, a, false, bf,
                                                  (short)0, acc, false, false);
  }

  // C/D layout: lane ln = col n0+ln, VGPR r -> row o0 + r + 8*hi
  if (g < 4) {
    __bf16* dst = (__bf16*)((g < 2) ? qbuf : kbuf);
    const int obase = (g & 1) * 16 + 8 * hi;
    dst += (size_t)b * NN * DD + (size_t)(n0 + ln) * DD + obase;
    const float* bp = bias + (g & 1) * 16 + 8 * hi - (g & 1) * 16 + o0 - o0; // = bias + 8*hi
    #pragma unroll
    for (int r = 0; r < 8; ++r)
      dst[r] = (__bf16)(acc[r] + bias[(g & 1) * 16 + 8 * hi + r - (g & 1) * 16 + (o0 & 16)]);
    (void)bp;
  } else {
    __bf16* dst = (__bf16*)vbuf + (size_t)b * CC * NN
                + (size_t)(o0 + 8 * hi) * NN + n0 + ln;
    #pragma unroll
    for (int r = 0; r < 8; ++r)
      dst[(size_t)r * NN] = (__bf16)(acc[r] + bias[o0 + 8 * hi + r]);
  }
}

// ---------------------------------------------------------------------------
// TDM: load a JTx32 bf16 tile (rows j0..j0+JT-1 of the [N,32] k tensor)
// into LDS at byte offset ldsOff. 2D descriptor, data_size=2B.
// ---------------------------------------------------------------------------
static __device__ __forceinline__ void tdm_load_k_tile(const void* gptr,
                                                       unsigned ldsOff) {
  unsigned long long ga = (unsigned long long)(uintptr_t)gptr;
  v4u g0;
  g0[0] = 1u;                                  // count=1, user descriptor
  g0[1] = ldsOff;                              // lds_addr (bytes)
  g0[2] = (unsigned)ga;                        // global_addr[31:0]
  g0[3] = (unsigned)(ga >> 32) | (2u << 30);   // global_addr[56:32] | type=2
  v8i g1;
  g1[0] = 0x00010000;                          // data_size=1 (2 bytes)
  g1[1] = (int)(DD << 16);                     // tensor_dim0 = 32
  g1[2] = (int)(NN << 16);                     // tensor_dim1 = 4096
  g1[3] = (int)(DD << 16);                     // tile_dim0 = 32
  g1[4] = (int)JT;                             // tile_dim1 = 64
  g1[5] = (int)DD;                             // tensor_dim0_stride = 32
  g1[6] = 0;
  g1[7] = 0;
  v4i z = {0, 0, 0, 0};                        // groups 2/3 unused (2D tensor)
#if defined(__clang_major__) && (__clang_major__ >= 23)
  v8i z8 = {0, 0, 0, 0, 0, 0, 0, 0};
  __builtin_amdgcn_tensor_load_to_lds(g0, g1, z, z, z8, 0);
#else
  __builtin_amdgcn_tensor_load_to_lds(g0, g1, z, z, 0);
#endif
}

// ---------------------------------------------------------------------------
// Kernel 2: flash attention, transposed-S formulation.
//   grid = (N/16, B), block = 64 threads (2 waves); wave w owns channels
//   [128w, 128w+128) = 8 accumulator c-tiles.
//   Per JT=64 j-step: TDM K tile -> LDS (double-buffered, wave 0 issues),
//   4x S^T WMMA, online softmax (per-lane + one shfl_xor(16)), 16x O^T WMMA.
// ---------------------------------------------------------------------------
__global__ __launch_bounds__(64) void attn_kernel(
    const float* __restrict__ x,
    const unsigned short* __restrict__ qbuf,
    const unsigned short* __restrict__ kbuf,
    const unsigned short* __restrict__ vbuf,
    const float* __restrict__ gamma,
    float* __restrict__ out)
{
  __shared__ __align__(16) __bf16 ldsK[2][JT * DD];   // 2 x 4 KB

  const int lane = threadIdx.x & 31;
  const int wave = threadIdx.x >> 5;
  const int ln = lane & 15;
  const int hi = lane >> 4;
  const int i0 = blockIdx.x * 16;
  const int b  = blockIdx.y;

  const __bf16* qb = (const __bf16*)qbuf + (size_t)b * NN * DD;
  const __bf16* kb = (const __bf16*)kbuf + (size_t)b * NN * DD;
  const __bf16* vb = (const __bf16*)vbuf + (size_t)b * CC * NN;

  const unsigned ldsBase0 = (unsigned)(uintptr_t)(void*)&ldsK[0][0];
  const unsigned ldsBase1 = (unsigned)(uintptr_t)(void*)&ldsK[1][0];

  // Q^T B-frag: lane column i = i0+ln, K(d) halves by hi (loaded once)
  v16bf qf;
  {
    const __bf16* qp = qb + (size_t)(i0 + ln) * DD + hi * 8;
    ((v8bf*)&qf)[0] = *(const v8bf*)qp;
    ((v8bf*)&qf)[1] = *(const v8bf*)(qp + 16);
  }

  v8f oacc[8] = {};
  float m = -INFINITY, l = 0.0f;

  const __bf16* vcol0 = vb + (size_t)(wave * 128 + ln) * NN + hi * 8;

  const int T = NN / JT;
  if (wave == 0) tdm_load_k_tile(kb, ldsBase0);      // prologue: tile 0 -> buf 0

  for (int t = 0; t < T; ++t) {
    const int j0 = t * JT;
    __syncthreads();                 // everyone done reading buf[(t+1)&1]
    if (wave == 0) {
      if (t + 1 < T) {
        tdm_load_k_tile(kb + (size_t)(j0 + JT) * DD,
                        ((t + 1) & 1) ? ldsBase1 : ldsBase0);
        __builtin_amdgcn_s_wait_tensorcnt(1);   // tile t landed, t+1 in flight
      } else {
        __builtin_amdgcn_s_wait_tensorcnt(0);
      }
    }
    __syncthreads();                 // buf[t&1] visible to both waves
    const __bf16* kt = &ldsK[t & 1][0];

    // --- K A-frags from LDS, 4 x S^T = K x Q^T ---
    v8f s[4];
    #pragma unroll
    for (int jj = 0; jj < 4; ++jj) {
      const __bf16* kr = kt + (size_t)(jj * 16 + ln) * DD + hi * 8;
      v16bf ka;
      ((v8bf*)&ka)[0] = *(const v8bf*)kr;        // ds_load_b128
      ((v8bf*)&ka)[1] = *(const v8bf*)(kr + 16);
      v8f zz = {};
      s[jj] = __builtin_amdgcn_wmma_f32_16x16x32_bf16(false, ka, false, qf,
                                                      (short)0, zz, false, false);
    }

    // --- online softmax over the 32 j's this lane holds ---
    float mx = s[0][0];
    #pragma unroll
    for (int jj = 0; jj < 4; ++jj)
      #pragma unroll
      for (int e = 0; e < 8; ++e) mx = fmaxf(mx, s[jj][e]);
    mx = fmaxf(mx, __shfl_xor(mx, 16, 32));
    const float mnew = fmaxf(m, mx);
    const float alpha = __expf(m - mnew);

    // P^T B-frags: frag f covers j-chunk 32f..32f+31; elements 0..7 <- tile 2f,
    // 8..15 <- tile 2f+1. No cross-lane movement (S^T C/D layout == B layout).
    v16bf pt[2];
    float ssum = 0.0f;
    #pragma unroll
    for (int f = 0; f < 2; ++f) {
      #pragma unroll
      for (int e = 0; e < 8; ++e) {
        float p = __expf(s[2 * f][e] - mnew);     ssum += p; pt[f][e]     = (__bf16)p;
        float q = __expf(s[2 * f + 1][e] - mnew); ssum += q; pt[f][8 + e] = (__bf16)q;
      }
    }
    ssum += __shfl_xor(ssum, 16, 32);
    l = l * alpha + ssum;
    m = mnew;

    // rescale accumulator (alpha is per-lane: pure VALU splat)
    #pragma unroll
    for (int t2 = 0; t2 < 8; ++t2)
      #pragma unroll
      for (int e = 0; e < 8; ++e) oacc[t2][e] *= alpha;

    // --- O^T += V x P^T, 8 channel tiles x 2 j-chunks ---
    #pragma unroll
    for (int t2 = 0; t2 < 8; ++t2) {
      const __bf16* vp = vcol0 + (size_t)(t2 * 16) * NN + j0;
      #pragma unroll
      for (int f = 0; f < 2; ++f) {
        v16bf vf;
        ((v8bf*)&vf)[0] = *(const v8bf*)(vp + 32 * f);
        ((v8bf*)&vf)[1] = *(const v8bf*)(vp + 32 * f + 16);
        oacc[t2] = __builtin_amdgcn_wmma_f32_16x16x32_bf16(false, vf, false, pt[f],
                                                           (short)0, oacc[t2], false, false);
      }
    }
  }

  // --- epilogue: out = gamma * O / l + x ---
  const float gl = gamma[0] / l;
  float* ob = out + (size_t)b * CC * NN;
  const float* xb = x + (size_t)b * CC * NN;
  #pragma unroll
  for (int t2 = 0; t2 < 8; ++t2) {
    const int cbase = wave * 128 + t2 * 16 + hi * 8;
    size_t idx = (size_t)cbase * NN + i0 + ln;
    #pragma unroll
    for (int r = 0; r < 8; ++r) {
      const size_t id = idx + (size_t)r * NN;
      ob[id] = gl * oacc[t2][r] + xb[id];
    }
  }
}

// ---------------------------------------------------------------------------
extern "C" void kernel_launch(void* const* d_in, const int* in_sizes, int n_in,
                              void* d_out, int out_size, void* d_ws, size_t ws_size,
                              hipStream_t stream) {
  const float* x     = (const float*)d_in[0];
  const float* Wq    = (const float*)d_in[1];
  const float* bq    = (const float*)d_in[2];
  const float* Wk    = (const float*)d_in[3];
  const float* bk    = (const float*)d_in[4];
  const float* Wv    = (const float*)d_in[5];
  const float* bv    = (const float*)d_in[6];
  const float* gamma = (const float*)d_in[7];
  float* out = (float*)d_out;

  char* ws = (char*)d_ws;
  const size_t qk_bytes = (size_t)BB * NN * DD * 2;   // 1 MB each
  unsigned short* qbuf = (unsigned short*)(ws);
  unsigned short* kbuf = (unsigned short*)(ws + qk_bytes);
  unsigned short* vbuf = (unsigned short*)(ws + 2 * qk_bytes);  // 8 MB

  dim3 g1(NN / 16, 20, BB);
  proj_kernel<<<g1, 32, 0, stream>>>(x, Wq, bq, Wk, bk, Wv, bv, qbuf, kbuf, vbuf);

  dim3 g2(NN / 16, BB);
  attn_kernel<<<g2, 64, 0, stream>>>(x, qbuf, kbuf, vbuf, gamma, out);
}
